// LSTMDecoder_68839735820941
// MI455X (gfx1250) — compile-verified, matrix operand order checked by reading, not run on hardware
//
#include <hip/hip_runtime.h>

// ---------------------------------------------------------------------------
// LSTM decoder (teacher_forcing=0) for MI455X / gfx1250.
// bf16 WMMA (v_wmma_f32_16x16x32_bf16) + fp32 accumulation.
// B-fragments staged through LDS with global_load_async_to_lds_b128
// (8KB double-buffered, ASYNCcnt-synchronized, 2 K-chunks per barrier)
// and shared by 8 waves/workgroup. Packed bf16 weights (36 MB) stay resident
// in the 192 MB L2 across all 128 sequential timesteps.
// ---------------------------------------------------------------------------

#define B_   512
#define T_   128
#define D_   128
#define H_   2048
#define KXH  (D_ + H_)     // 2176 combined K for gates GEMM ([x | h])
#define KC_G (KXH / 32)    // 68 K-chunks of 32 for gates (multiple of 4)
#define KC_O (H_ / 32)     // 64 K-chunks for output projection
#define GSTRIDE ((size_t)128 * KC_G * 256)   // dwords per gate plane in wg_p

typedef __attribute__((ext_vector_type(16))) __bf16        bf16x16;
typedef __attribute__((ext_vector_type(8)))  float         f32x8;
typedef __attribute__((ext_vector_type(4)))  unsigned int  u32x4;

union FragU { u32x4 u[2]; bf16x16 v; };

__device__ __forceinline__ unsigned short f2bf(float f) {
    unsigned int u = __builtin_bit_cast(unsigned int, f);
    unsigned int r = u + 0x7FFFu + ((u >> 16) & 1u);   // round-to-nearest-even
    return (unsigned short)(r >> 16);
}
__device__ __forceinline__ float fsig(float x) {
    return 1.0f / (1.0f + __expf(-x));
}
__device__ __forceinline__ float ftanh(float x) {
    float e = __expf(2.0f * x);
    return 1.0f - 2.0f / (e + 1.0f);   // robust at +/-inf
}

// Async copy of 16B/lane from global to LDS (ASYNCcnt-tracked).
__device__ __forceinline__ void async_g2l_b128(unsigned lds_addr, const void* gaddr) {
    asm volatile("global_load_async_to_lds_b128 %0, %1, off"
                 :: "v"(lds_addr), "v"((unsigned long long)(size_t)gaddr)
                 : "memory");
}
__device__ __forceinline__ void wait_asynccnt0() {
    asm volatile("s_wait_asynccnt 0" ::: "memory");
}

// Load one A fragment (16x32 bf16, this wave's batch rows) from global.
__device__ __forceinline__ FragU loadA(const unsigned short* arow, int kc) {
    FragU a;
    const u32x4* ap = (const u32x4*)(arow + kc * 32);
    a.u[0] = ap[0];
    a.u[1] = ap[2];
    return a;
}

// 4 gate WMMAs from one LDS-staged 4KB B block.
__device__ __forceinline__ void wmma4(const unsigned int* lbase, const FragU& a, f32x8* acc) {
    #pragma unroll
    for (int g = 0; g < 4; ++g) {
        FragU b;
        const u32x4* bp = (const u32x4*)(lbase + g * 256);
        b.u[0] = bp[0];
        b.u[1] = bp[1];
        acc[g] = __builtin_amdgcn_wmma_f32_16x16x32_bf16(
            false, a.v, false, b.v, (short)0, acc[g], false, false);
    }
}

// B-fragment lane mapping for 32(K) x 16(N) bf16 tile (mirror of ISA A layout)
__device__ __forceinline__ int frag_k0(int lane, int v) {
    return ((v < 4) ? 0 : 16) + 8 * (lane >> 4) + 2 * (v & 3);
}

// -------------------------- one-time packing -------------------------------

// Wg packed layout (dwords): [gate 4][jt 128][kc 68][lane 32][v 8]
__global__ void pack_wg(const float* __restrict__ Wih, const float* __restrict__ Whh,
                        unsigned int* __restrict__ dst) {
    int id = blockIdx.x * 256 + threadIdx.x;            // 8,912,896 total
    int v    = id & 7;        int rest = id >> 3;
    int lane = rest & 31;     rest >>= 5;
    int kc   = rest % KC_G;   rest /= KC_G;
    int jt   = rest & 127;    int g = rest >> 7;
    int n  = g * H_ + jt * 16 + (lane & 15);
    int k0 = kc * 32 + frag_k0(lane, v);
    float w0 = (k0     < D_) ? Wih[(size_t)n * D_ + k0]     : Whh[(size_t)n * H_ + (k0     - D_)];
    float w1 = (k0 + 1 < D_) ? Wih[(size_t)n * D_ + k0 + 1] : Whh[(size_t)n * H_ + (k0 + 1 - D_)];
    dst[id] = (unsigned int)f2bf(w0) | ((unsigned int)f2bf(w1) << 16);
}

// Wout packed layout (dwords): [nt 8][kc 64][lane 32][v 8]
__global__ void pack_wout(const float* __restrict__ Wout, unsigned int* __restrict__ dst) {
    int id = blockIdx.x * 256 + threadIdx.x;            // 131,072 total
    int v    = id & 7;         int rest = id >> 3;
    int lane = rest & 31;      rest >>= 5;
    int kc   = rest & 63;      int nt = rest >> 6;
    int n  = nt * 16 + (lane & 15);
    int k0 = kc * 32 + frag_k0(lane, v);
    dst[id] = (unsigned int)f2bf(Wout[(size_t)n * H_ + k0]) |
              ((unsigned int)f2bf(Wout[(size_t)n * H_ + k0 + 1]) << 16);
}

__global__ void fuse_bias(const float* __restrict__ bih, const float* __restrict__ bhh,
                          float* __restrict__ bias) {
    int i = blockIdx.x * 256 + threadIdx.x;
    if (i < 4 * H_) bias[i] = bih[i] + bhh[i];
}

// xh0[:, :D] = bf16(tgt[:,0,:]) ; xh0[:, D:] = 0 ; c = 0
__global__ void init_state(const float* __restrict__ tgt,
                           unsigned short* __restrict__ xh0, float* __restrict__ c) {
    int id = blockIdx.x * 256 + threadIdx.x;            // B_*KXH threads
    if (id < B_ * KXH) {
        int b = id / KXH, k = id % KXH;
        xh0[id] = (k < D_) ? f2bf(tgt[(size_t)b * T_ * D_ + k]) : (unsigned short)0;
    }
    if (id < B_ * H_) c[id] = 0.0f;
}

// ------------------------------ main step ----------------------------------

// Workgroup = 8 waves, all with the same hidden tile jt, batch tiles
// mt = mgroup*8 + wave. Per async round the workgroup stages an 8KB block
// (2 K-chunks x 4 gates x 1KB fragment) into LDS: each thread async-copies
// 16B into each 4KB half, then s_wait_asynccnt + barrier. 8 WMMAs per
// barrier, double-buffered, with A-fragments register-pipelined from global.
// Explicit even/odd phases keep fragments/accumulators pinned (no phi moves).
__global__ void __launch_bounds__(256)
lstm_gates(const unsigned short* __restrict__ xh_in,   // [B, 2176] bf16 (x|h) step t
           unsigned short* __restrict__ xh_out,        // h-part written (step t+1 buffer)
           const unsigned int* __restrict__ wg,        // packed bf16 weights
           const float* __restrict__ bias,             // fused b_ih+b_hh [8192]
           float* __restrict__ c)                      // [B, H] fp32 cell state
{
    __shared__ unsigned int bstage[2][2048];           // 2 x 8KB double buffer

    const int tid  = threadIdx.x;
    const int wave = tid >> 5;
    const int lane = tid & 31;
    const int jt     = blockIdx.x & 127;               // hidden tile
    const int mgroup = blockIdx.x >> 7;                // 0..3
    const int mt     = mgroup * 8 + wave;              // batch tile 0..31
    const int half = lane >> 4;
    const int nl   = lane & 15;
    const int jcol = jt * 16 + nl;

    // --- async-copy segment owned by this thread: 16B within each 4KB half
    const int sb = tid * 16;                           // 0..4095
    const unsigned int* gsrc = wg + (size_t)jt * KC_G * 256
                                  + (size_t)(sb >> 10) * GSTRIDE + ((sb & 1023) >> 2);
    const unsigned ldsb0 = (unsigned)(size_t)&bstage[0][0];
    const unsigned ldsb1 = (unsigned)(size_t)&bstage[1][0];

    // --- accumulators init with fused bias
    f32x8 acc[4];
    #pragma unroll
    for (int g = 0; g < 4; ++g) {
        float bv = bias[g * H_ + jcol];
        #pragma unroll
        for (int r = 0; r < 8; ++r) acc[g][r] = bv;
    }

    const unsigned short* arow = xh_in + (size_t)(mt * 16 + nl) * KXH + 8 * half;

    // --- prologue: stage K-chunks {0,1} into buf0, prefetch A(0), A(1)
    async_g2l_b128(ldsb0 + sb,        gsrc);
    async_g2l_b128(ldsb0 + 4096 + sb, gsrc + 256);
    FragU a0 = loadA(arow, 0);
    FragU a1 = loadA(arow, 1);
    wait_asynccnt0();
    __syncthreads();

    FragU a2, a3;
    for (int kp = 0; kp < KC_G; kp += 4) {
        // ---- phase 0: consume buf0 {kp, kp+1}; stage {kp+2, kp+3} -> buf1
        async_g2l_b128(ldsb1 + sb,        gsrc + (size_t)(kp + 2) * 256);
        async_g2l_b128(ldsb1 + 4096 + sb, gsrc + (size_t)(kp + 3) * 256);
        a2 = loadA(arow, kp + 2);
        a3 = loadA(arow, kp + 3);
        wmma4(&bstage[0][lane * 8],        a0, acc);
        wmma4(&bstage[0][1024 + lane * 8], a1, acc);
        wait_asynccnt0();
        __syncthreads();
        // ---- phase 1: consume buf1 {kp+2, kp+3}; stage {kp+4, kp+5} -> buf0
        const bool more = (kp + 4) < KC_G;
        if (more) {
            async_g2l_b128(ldsb0 + sb,        gsrc + (size_t)(kp + 4) * 256);
            async_g2l_b128(ldsb0 + 4096 + sb, gsrc + (size_t)(kp + 5) * 256);
            a0 = loadA(arow, kp + 4);
            a1 = loadA(arow, kp + 5);
        }
        wmma4(&bstage[1][lane * 8],        a2, acc);
        wmma4(&bstage[1][1024 + lane * 8], a3, acc);
        if (more) {
            wait_asynccnt0();
            __syncthreads();
        }
    }

    // --- LSTM cell update (fp32), write new h as bf16 into ping-pong buffer
    const int brow0 = mt * 16 + 8 * half;
    #pragma unroll
    for (int r = 0; r < 8; ++r) {
        int b = brow0 + r;
        float iv = fsig(acc[0][r]);
        float fv = fsig(acc[1][r]);
        float gv = ftanh(acc[2][r]);
        float ov = fsig(acc[3][r]);
        size_t ci = (size_t)b * H_ + jcol;
        float cn = fv * c[ci] + iv * gv;
        c[ci] = cn;
        xh_out[(size_t)b * KXH + D_ + jcol] = f2bf(ov * ftanh(cn));
    }
}

// out = h @ Wout^T + b_out ; writes fp32 to d_out[:,t,:] and bf16 next-x.
// Explicit even/odd 2-deep register pipeline (loads issued one full
// iteration ahead of the consuming WMMA, no phi copies).
__global__ void __launch_bounds__(256)
lstm_out(unsigned short* __restrict__ xh,              // reads h-part, writes x-part
         const unsigned int* __restrict__ wout,
         const float* __restrict__ bout,
         float* __restrict__ out, int t)
{
    const int wid  = (blockIdx.x * 256 + threadIdx.x) >> 5;
    const int lane = threadIdx.x & 31;
    const int mt   = wid >> 3;        // batch tile 0..31
    const int nt   = wid & 7;         // out-dim tile 0..7
    const int half = lane >> 4;
    const int nl   = lane & 15;

    f32x8 acc;
    {
        float bv = bout[nt * 16 + nl];
        #pragma unroll
        for (int r = 0; r < 8; ++r) acc[r] = bv;
    }

    const unsigned short* arow  = xh + (size_t)(mt * 16 + nl) * KXH + D_ + 8 * half;
    const unsigned int*   bbase = wout + (size_t)nt * KC_O * 256 + lane * 8;

    FragU a0, a1, b0, b1;
    a0 = loadA(arow, 0);
    {
        const u32x4* bp = (const u32x4*)(bbase);
        b0.u[0] = bp[0];  b0.u[1] = bp[1];
    }
    for (int kc = 0; kc < KC_O; kc += 2) {
        a1 = loadA(arow, kc + 1);
        {
            const u32x4* bp = (const u32x4*)(bbase + (size_t)(kc + 1) * 256);
            b1.u[0] = bp[0];  b1.u[1] = bp[1];
        }
        acc = __builtin_amdgcn_wmma_f32_16x16x32_bf16(
            false, a0.v, false, b0.v, (short)0, acc, false, false);
        if (kc + 2 < KC_O) {
            a0 = loadA(arow, kc + 2);
            const u32x4* bp = (const u32x4*)(bbase + (size_t)(kc + 2) * 256);
            b0.u[0] = bp[0];  b0.u[1] = bp[1];
        }
        acc = __builtin_amdgcn_wmma_f32_16x16x32_bf16(
            false, a1.v, false, b1.v, (short)0, acc, false, false);
    }

    const int n     = nt * 16 + nl;
    const int brow0 = mt * 16 + 8 * half;
    #pragma unroll
    for (int r = 0; r < 8; ++r) {
        int bb = brow0 + r;
        out[((size_t)bb * T_ + t) * D_ + n] = acc[r];
        xh[(size_t)bb * KXH + n] = f2bf(acc[r]);   // next-step input
    }
}

// ------------------------------- launch ------------------------------------

extern "C" void kernel_launch(void* const* d_in, const int* in_sizes, int n_in,
                              void* d_out, int out_size, void* d_ws, size_t ws_size,
                              hipStream_t stream) {
    (void)in_sizes; (void)n_in; (void)out_size; (void)ws_size;
    const float* tgt  = (const float*)d_in[0];
    const float* Wih  = (const float*)d_in[1];
    const float* Whh  = (const float*)d_in[2];
    const float* bih  = (const float*)d_in[3];
    const float* bhh  = (const float*)d_in[4];
    const float* Wout = (const float*)d_in[5];
    const float* bout = (const float*)d_in[6];
    float* out = (float*)d_out;

    char* ws = (char*)d_ws;
    unsigned int*   wg_p   = (unsigned int*)(ws);                 // 35,651,584 B
    unsigned int*   wout_p = (unsigned int*)(ws + 35651584);      //    524,288 B
    float*          bias   = (float*)(ws + 36175872);             //     32,768 B
    unsigned short* xh0    = (unsigned short*)(ws + 36208640);    //  2,228,224 B
    unsigned short* xh1    = (unsigned short*)(ws + 38436864);    //  2,228,224 B
    float*          cbuf   = (float*)(ws + 40665088);             //  4,194,304 B
    // total ws usage: 44,859,392 B

    pack_wg   <<<34816, 256, 0, stream>>>(Wih, Whh, wg_p);
    pack_wout <<<  512, 256, 0, stream>>>(Wout, wout_p);
    fuse_bias <<<   32, 256, 0, stream>>>(bih, bhh, bias);
    init_state<<< 4352, 256, 0, stream>>>(tgt, xh0, cbuf);

    for (int t = 0; t < T_; ++t) {
        const unsigned short* xin = (t & 1) ? xh1 : xh0;   // [x_t | h_{t-1}]
        unsigned short* xnext     = (t & 1) ? xh0 : xh1;   // receives h_t then x_{t+1}
        lstm_gates<<<512, 256, 0, stream>>>(xin, xnext, wg_p, bias, cbuf);
        lstm_out  <<< 32, 256, 0, stream>>>(xnext, wout_p, bout, out, t);
    }
}
